// ForceMapping_63891933495827
// MI455X (gfx1250) — compile-verified, matrix operand order checked by reading, not run on hardware
//
#include <hip/hip_runtime.h>
#include <stdint.h>

typedef _Float16 v16h __attribute__((ext_vector_type(16)));
typedef _Float16 v8h  __attribute__((ext_vector_type(8)));
typedef float    v8f  __attribute__((ext_vector_type(8)));
typedef float    v4f  __attribute__((ext_vector_type(4)));

#define NBR 64
#define FD  128
#define HD  64

__device__ __forceinline__ float ssp(float x) {
    // softplus(x) - log(2) via hardware transcendentals:
    // softplus(x) = max(x,0) + log(1 + exp(-|x|))
    return fmaxf(x, 0.0f) + __logf(1.0f + __expf(-fabsf(x))) - 0.69314718055994531f;
}

extern "C" __global__ __launch_bounds__(128)
void force_map_kernel(const float* __restrict__ E,    // (BA, 64, 128)
                      const float* __restrict__ UV,   // (BA, 64, 3)
                      const float* __restrict__ W1,   // (128, 64)
                      const float* __restrict__ B1,   // (64)
                      const float* __restrict__ W2,   // (64)
                      const float* __restrict__ B2,   // (1)
                      float* __restrict__ out)        // (BA, 3)
{
    __shared__ float    tileE[NBR * FD];   // 32 KB : E tile (f32)
    __shared__ _Float16 w1t[HD * FD];      // 16 KB : W1 transposed [n][k] as f16
    __shared__ float    fmbuf[NBR];        // per-neighbor force magnitude

    const int tid  = threadIdx.x;
    const int lane = tid & 31;
    const int wave = tid >> 5;
    const int col  = lane & 15;            // N / M sub-index within 16-wide tile
    const int khalf = (lane >> 4) << 3;    // K offset 0 or 8 (16-bit A/B layout)
    const int ba   = blockIdx.x;

    // ---- async copy of the 64x128 f32 E tile into LDS (2048 x b128, streaming) ----
    {
        const float* src = E + (size_t)ba * (NBR * FD);
        uint32_t lds_base = (uint32_t)(uintptr_t)(&tileE[0]);
        #pragma unroll
        for (int i = 0; i < 16; ++i) {
            uint32_t goff  = (uint32_t)((i * 128 + tid) * 16);
            uint32_t laddr = lds_base + goff;
            asm volatile("global_load_async_to_lds_b128 %0, %1, %2 th:TH_LOAD_NT"
                         :: "v"(laddr), "v"(goff), "s"(src) : "memory");
        }
    }

    // ---- stage W1 as f16, transposed (vectorized: 16 x b128 per thread) ----
    #pragma unroll 4
    for (int c = tid; c < (FD * HD) / 4; c += 128) {
        const int base = c * 4;
        const int k = base >> 6;           // /HD
        const int n = base & 63;           // %HD (row-aligned: 64 % 4 == 0)
        v4f w = *(const v4f*)(W1 + base);
        w1t[(n + 0) * FD + k] = (_Float16)w.x;
        w1t[(n + 1) * FD + k] = (_Float16)w.y;
        w1t[(n + 2) * FD + k] = (_Float16)w.z;
        w1t[(n + 3) * FD + k] = (_Float16)w.w;
    }

    asm volatile("s_wait_asynccnt 0x0" ::: "memory");
    __syncthreads();

    // ---- GEMM: (64x128) x (128x64), wave w owns rows [16w, 16w+16) ----
    v8f acc[4] = {};                               // 4 n-tiles of 16x16 f32
    const float* arow = &tileE[(wave * 16 + col) * FD];

    #pragma unroll
    for (int k = 0; k < 4; ++k) {                  // K step = 32
        const int koff = k * 32 + khalf;
        // A fragment: halves 0..7 = K[koff..koff+7], 8..15 = K[koff+16..koff+23]
        v4f a0 = *(const v4f*)(arow + koff);
        v4f a1 = *(const v4f*)(arow + koff + 4);
        v4f a2 = *(const v4f*)(arow + koff + 16);
        v4f a3 = *(const v4f*)(arow + koff + 20);
        v16h a;
        #pragma unroll
        for (int i = 0; i < 4; ++i) {
            a[i]      = (_Float16)a0[i];
            a[4 + i]  = (_Float16)a1[i];
            a[8 + i]  = (_Float16)a2[i];
            a[12 + i] = (_Float16)a3[i];
        }
        #pragma unroll
        for (int nt = 0; nt < 4; ++nt) {
            const _Float16* bcol = &w1t[(nt * 16 + col) * FD + koff];
            v8h bl = *(const v8h*)(bcol);
            v8h bh = *(const v8h*)(bcol + 16);
            v16h b = __builtin_shufflevector(bl, bh,
                       0,1,2,3,4,5,6,7,8,9,10,11,12,13,14,15);
            acc[nt] = __builtin_amdgcn_wmma_f32_16x16x32_f16(
                          false, a, false, b, (short)0, acc[nt], false, false);
        }
    }

    // ---- layer 2: fm = ssp(h + b1) . W2 + b2, reduced across the 16 lanes/row ----
    float b1v[4], w2v[4];
    #pragma unroll
    for (int nt = 0; nt < 4; ++nt) {
        b1v[nt] = B1[nt * 16 + col];
        w2v[nt] = W2[nt * 16 + col];
    }
    const float bias2 = B2[0];

    #pragma unroll
    for (int v = 0; v < 8; ++v) {
        float s = 0.0f;
        #pragma unroll
        for (int nt = 0; nt < 4; ++nt)
            s += ssp(acc[nt][v] + b1v[nt]) * w2v[nt];
        // reduce within each 16-lane half (masks < 16 stay inside the half)
        s += __shfl_xor(s, 1);
        s += __shfl_xor(s, 2);
        s += __shfl_xor(s, 4);
        s += __shfl_xor(s, 8);
        if (col == 0) {
            int row = wave * 16 + v + ((lane >> 4) << 3);  // C layout: M=v / v+8
            fmbuf[row] = s + bias2;
        }
    }
    __syncthreads();

    // ---- fold 64 neighbors: out[ba][c] = sum_n fm[n] * uv[n][c] ----
    if (wave == 0) {
        const float* uv = UV + (size_t)ba * (NBR * 3);
        float f0 = fmbuf[lane], f1 = fmbuf[lane + 32];
        float p0 = f0 * uv[lane * 3 + 0] + f1 * uv[(lane + 32) * 3 + 0];
        float p1 = f0 * uv[lane * 3 + 1] + f1 * uv[(lane + 32) * 3 + 1];
        float p2 = f0 * uv[lane * 3 + 2] + f1 * uv[(lane + 32) * 3 + 2];
        #pragma unroll
        for (int m = 1; m < 32; m <<= 1) {
            p0 += __shfl_xor(p0, m);
            p1 += __shfl_xor(p1, m);
            p2 += __shfl_xor(p2, m);
        }
        if (lane == 0) {
            out[(size_t)ba * 3 + 0] = p0;
            out[(size_t)ba * 3 + 1] = p1;
            out[(size_t)ba * 3 + 2] = p2;
        }
    }
}

extern "C" void kernel_launch(void* const* d_in, const int* in_sizes, int n_in,
                              void* d_out, int out_size, void* d_ws, size_t ws_size,
                              hipStream_t stream) {
    (void)n_in; (void)out_size; (void)d_ws; (void)ws_size;
    const float* E  = (const float*)d_in[0];
    const float* UV = (const float*)d_in[1];
    const float* W1 = (const float*)d_in[2];
    const float* B1 = (const float*)d_in[3];
    const float* W2 = (const float*)d_in[4];
    const float* B2 = (const float*)d_in[5];
    float* out = (float*)d_out;

    const int BA = in_sizes[0] / (NBR * FD);   // 16*512 = 8192 atom slots
    force_map_kernel<<<BA, 128, 0, stream>>>(E, UV, W1, B1, W2, B2, out);
}
